// ngcf3_session_hot_items_88957362635444
// MI455X (gfx1250) — compile-verified, matrix operand order checked by reading, not run on hardware
//
#include <hip/hip_runtime.h>

// ---------------- problem constants (from reference) ----------------
#define N_SESS   50000
#define N_ITEMS  100000
#define NTOT     150000          // N_SESS + N_ITEMS
#define EMB      128
#define HH       128             // H1 == H2
#define H3DIM    64
#define A_NNZ    4800000
#define S_NNZ    1000000
#define BATCH    1024

typedef __attribute__((ext_vector_type(16))) __bf16 v16bf;
typedef __attribute__((ext_vector_type(8)))  float  v8f;

union BF16x16 { v16bf v; unsigned short u[16]; uint4 q[2]; };

__device__ __forceinline__ unsigned short f32_to_bf16(float f) {
    unsigned int x = __float_as_uint(f);
    unsigned int r = (x + 0x7FFFu + ((x >> 16) & 1u)) >> 16;   // RNE
    return (unsigned short)r;
}
__device__ __forceinline__ float bf16_to_f32(unsigned short h) {
    return __uint_as_float(((unsigned int)h) << 16);
}

// ---------------- elementwise helpers ----------------
__global__ void zero_f32_kernel(float* __restrict__ p, int n) {
    int i = blockIdx.x * blockDim.x + threadIdx.x;
    int stride = gridDim.x * blockDim.x;
    for (; i < n; i += stride) p[i] = 0.0f;
}

__global__ void cvt_f32_bf16_kernel(const float* __restrict__ src,
                                    unsigned short* __restrict__ dst, int n) {
    int i = blockIdx.x * blockDim.x + threadIdx.x;
    int stride = gridDim.x * blockDim.x;
    for (; i < n; i += stride) dst[i] = f32_to_bf16(src[i]);
}

// acc[N*d] + bias[d], optional relu, -> bf16   (d is a power of two)
__global__ void bias_relu_cvt_kernel(const float* __restrict__ acc,
                                     const float* __restrict__ bias,
                                     unsigned short* __restrict__ dst,
                                     int n, int dmask, int relu) {
    int i = blockIdx.x * blockDim.x + threadIdx.x;
    int stride = gridDim.x * blockDim.x;
    for (; i < n; i += stride) {
        float v = acc[i] + bias[i & dmask];
        if (relu) v = fmaxf(v, 0.0f);
        dst[i] = f32_to_bf16(v);
    }
}

// ---------------- W fragment pre-pack ----------------
// Wp[(((kk/32)*(Kout/16) + tileN)*32 + lane)*16 + j] = bf16(W[k][tileN*16 + lane%16])
// per-lane K pattern of v_wmma_f32_16x16x32_bf16 B operand:
//   pair p = j>>1, hi = j&1, half = lane>>4
//   k = kk + 8*half + (p<4 ? 2*p+hi : 16 + 2*(p-4)+hi)
__global__ void pack_W_kernel(const float* __restrict__ W,
                              unsigned short* __restrict__ Wp,
                              int Kin, int Kout) {
    int idx = blockIdx.x * blockDim.x + threadIdx.x;
    int total = Kin * Kout;
    if (idx >= total) return;
    int j     = idx & 15;
    int lane  = (idx >> 4) & 31;
    int nt    = Kout >> 4;
    int tileN = (idx >> 9) % nt;
    int kk32  = idx / (512 * nt);
    int half = lane >> 4, lm = lane & 15;
    int p = j >> 1, hi = j & 1;
    int k = kk32 * 32 + 8 * half + ((p < 4) ? (2 * p + hi) : (16 + 2 * (p - 4) + hi));
    Wp[idx] = f32_to_bf16(W[k * Kout + tileN * 16 + lm]);
}

// ---------------- SpMM (COO scatter-add), multiple nnz per block -------------
// x_sess[r,k] += v * emb[c,k]   (d == 128, 2 nnz per 256-thread block)
__global__ void sess_spmm_kernel(const int* __restrict__ rows,
                                 const int* __restrict__ cols,
                                 const float* __restrict__ vals,
                                 const float* __restrict__ emb,
                                 float* __restrict__ acc, int nnz) {
    int e = blockIdx.x * 2 + (threadIdx.x >> 7);
    int k = threadIdx.x & 127;
    if (e >= nnz) return;
    int r = rows[e], c = cols[e];
    float v = vals[e];
    atomicAdd(&acc[r * EMB + k], v * emb[c * EMB + k]);
}

// acc[r,k] += v * bf16(t[c,k])   (t bf16 in L2; EPB nnz per 256-thread block)
template<int D>
__global__ void a_spmm_kernel(const int* __restrict__ rows,
                              const int* __restrict__ cols,
                              const float* __restrict__ vals,
                              const unsigned short* __restrict__ t,
                              float* __restrict__ acc, int nnz) {
    constexpr int EPB = 256 / D;
    int e = blockIdx.x * EPB + threadIdx.x / D;
    int k = threadIdx.x & (D - 1);
    if (e >= nnz) return;
    int r = rows[e], c = cols[e];
    float v = vals[e];
    float x = bf16_to_f32(t[c * D + k]);
    atomicAdd(&acc[r * D + k], v * x);
}

// ---------------- dense GEMM: T = H @ W, one wave per 16 x KOUT row block ----
// H: [n x Kin] bf16 row-major; Wp: fragment-packed bf16; T: [n x KOUT] bf16
template<int KOUT>
__global__ __launch_bounds__(32)
void gemm_wmma_rowblock(const unsigned short* __restrict__ H,
                        const unsigned short* __restrict__ Wp,
                        unsigned short* __restrict__ T, int Kin) {
    constexpr int NT = KOUT / 16;             // 16x16 tiles per row block
    const int tileM = blockIdx.x;
    const int lane  = threadIdx.x;
    const int half  = lane >> 4;
    const int lm    = lane & 15;

    v8f acc[NT];
    #pragma unroll
    for (int t = 0; t < NT; ++t) acc[t] = (v8f){};

    const unsigned short* hrow = H + (tileM * 16 + lm) * Kin;
    for (int kk = 0; kk < Kin; kk += 32) {
        // A fragment: two contiguous 16B runs (K = kk+8h..+7 and kk+16+8h..+7)
        BF16x16 a;
        const unsigned short* ap = hrow + kk + 8 * half;
        a.q[0] = *(const uint4*)(ap);
        a.q[1] = *(const uint4*)(ap + 16);
        const unsigned short* wb = Wp + ((kk >> 5) * NT * 32 + lane) * 16;
        #pragma unroll
        for (int t = 0; t < NT; ++t) {
            // B fragment: fully contiguous 32B per lane (pre-packed)
            BF16x16 b;
            const unsigned short* bp = wb + t * (32 * 16);
            b.q[0] = *(const uint4*)(bp);
            b.q[1] = *(const uint4*)(bp + 8);
            acc[t] = __builtin_amdgcn_wmma_f32_16x16x32_bf16(
                         false, a.v, false, b.v, (short)0, acc[t], false, false);
        }
    }
    // D: VGPR r holds (row = r + 8*half, col = lm) of each tile
    #pragma unroll
    for (int t = 0; t < NT; ++t) {
        #pragma unroll
        for (int r = 0; r < 8; ++r) {
            int orow = tileM * 16 + r + 8 * half;
            T[orow * KOUT + t * 16 + lm] = f32_to_bf16(acc[t][r]);
        }
    }
}

// ---------------- final: out[b,i] = h3[batch[b]] . h3[item i], K = 64 --------
// 410 MB write-once fp32 output dominates the whole pipeline at 23.3 TB/s
// -> non-temporal stores; math is trivial for WMMA.
__global__ __launch_bounds__(32)
void final_gemm_kernel(const unsigned short* __restrict__ H3,   // [NTOT x 64] bf16
                       const int* __restrict__ batch_idxes,     // [1024]
                       float* __restrict__ out) {               // [1024 x N_ITEMS]
    const int tileN = blockIdx.x;             // item tile (N_ITEMS/16)
    const int tileM = blockIdx.y;             // batch tile (BATCH/16)
    const int lane  = threadIdx.x;
    const int half  = lane >> 4;
    const int lm    = lane & 15;

    const int arow = batch_idxes[tileM * 16 + lm];      // session row of h3
    const int brow = N_SESS + tileN * 16 + lm;          // item row of h3

    v8f acc = {};
    #pragma unroll
    for (int kk = 0; kk < H3DIM; kk += 32) {
        BF16x16 a, b;
        const unsigned short* ap = H3 + arow * H3DIM + kk + 8 * half;
        a.q[0] = *(const uint4*)(ap);
        a.q[1] = *(const uint4*)(ap + 16);
        const unsigned short* bp = H3 + brow * H3DIM + kk + 8 * half;
        b.q[0] = *(const uint4*)(bp);
        b.q[1] = *(const uint4*)(bp + 16);
        acc = __builtin_amdgcn_wmma_f32_16x16x32_bf16(
                  false, a.v, false, b.v, (short)0, acc, false, false);
    }
    const int ocol = tileN * 16 + lm;
    #pragma unroll
    for (int r = 0; r < 8; ++r) {
        int orow = tileM * 16 + r + 8 * half;
        __builtin_nontemporal_store(acc[r], out + orow * N_ITEMS + ocol);
    }
}

// ---------------- launch ----------------
static inline int gsz(long n, int blk, long cap = 600000) {
    long g = (n + blk - 1) / blk;
    if (g > cap) g = cap;
    return (int)g;
}

extern "C" void kernel_launch(void* const* d_in, const int* in_sizes, int n_in,
                              void* d_out, int out_size, void* d_ws, size_t ws_size,
                              hipStream_t stream) {
    const int*   batch_idxes = (const int*)  d_in[0];
    const int*   A_rows      = (const int*)  d_in[1];
    const int*   A_cols      = (const int*)  d_in[2];
    const float* A_vals      = (const float*)d_in[3];
    /* item_idxes  d_in[4]  : identity (N_SESS + arange), folded into indexing */
    const int*   sess_rows   = (const int*)  d_in[5];
    const int*   sess_cols   = (const int*)  d_in[6];
    const float* sess_vals   = (const float*)d_in[7];
    /* item_emb_idxes d_in[8]: arange, folded into indexing */
    const float* emb         = (const float*)d_in[9];
    const float* W1 = (const float*)d_in[10];
    const float* b1 = (const float*)d_in[11];
    const float* W2 = (const float*)d_in[12];
    const float* b2 = (const float*)d_in[13];
    const float* W3 = (const float*)d_in[14];
    const float* b3 = (const float*)d_in[15];
    float* out = (float*)d_out;

    // workspace: hbuf (bf16 N x 128), tbuf (bf16 N x 128), acc (f32 N x 128), Wp
    const size_t HB  = (size_t)NTOT * 128 * sizeof(unsigned short);  // 38.4 MB
    const size_t ACC = (size_t)NTOT * 128 * sizeof(float);           // 76.8 MB
    char* ws = (char*)d_ws;
    unsigned short* hbuf = (unsigned short*)(ws);
    unsigned short* tbuf = (unsigned short*)(ws + HB);
    float*          acc  = (float*)         (ws + 2 * HB);
    unsigned short* Wp   = (unsigned short*)(ws + 2 * HB + ACC);     // 32 KB max

    // ---- session embedding aggregation: x_sess = S @ emb ----
    int nSessElems = N_SESS * EMB;
    zero_f32_kernel<<<gsz(nSessElems, 256), 256, 0, stream>>>(acc, nSessElems);
    sess_spmm_kernel<<<(S_NNZ + 1) / 2, 256, 0, stream>>>(
        sess_rows, sess_cols, sess_vals, emb, acc, S_NNZ);
    cvt_f32_bf16_kernel<<<gsz(nSessElems, 256), 256, 0, stream>>>(acc, hbuf, nSessElems);
    int nItemElems = N_ITEMS * EMB;
    cvt_f32_bf16_kernel<<<gsz(nItemElems, 256), 256, 0, stream>>>(
        emb, hbuf + N_SESS * EMB, nItemElems);

    // ---- 3 graph-conv layers: h = act(A @ (h @ W) + b) ----
    const float* Ws[3]    = {W1, W2, W3};
    const float* bs[3]    = {b1, b2, b3};
    const int    kouts[3] = {HH, HH, H3DIM};
    const int    relus[3] = {1, 1, 0};
    int kin = EMB;
    for (int l = 0; l < 3; ++l) {
        const int kout = kouts[l];
        // pack W into WMMA-fragment order (bf16), then t = h @ W via WMMA
        int wElems = kin * kout;
        pack_W_kernel<<<(wElems + 255) / 256, 256, 0, stream>>>(Ws[l], Wp, kin, kout);
        if (kout == HH)
            gemm_wmma_rowblock<HH><<<NTOT / 16, 32, 0, stream>>>(hbuf, Wp, tbuf, kin);
        else
            gemm_wmma_rowblock<H3DIM><<<NTOT / 16, 32, 0, stream>>>(hbuf, Wp, tbuf, kin);
        // acc = A @ t  (COO scatter, f32 atomics into L2-resident accumulator)
        int nElems = NTOT * kout;
        zero_f32_kernel<<<gsz(nElems, 256), 256, 0, stream>>>(acc, nElems);
        if (kout == HH)
            a_spmm_kernel<HH><<<(A_NNZ + 1) / 2, 256, 0, stream>>>(
                A_rows, A_cols, A_vals, tbuf, acc, A_NNZ);
        else
            a_spmm_kernel<H3DIM><<<(A_NNZ + 3) / 4, 256, 0, stream>>>(
                A_rows, A_cols, A_vals, tbuf, acc, A_NNZ);
        // h = act(acc + b) -> bf16
        bias_relu_cvt_kernel<<<gsz(nElems, 256), 256, 0, stream>>>(
            acc, bs[l], hbuf, nElems, kout - 1, relus[l]);
        kin = kout;
    }

    // ---- out = h3[batch] @ h3[items].T  (WMMA, non-temporal f32 stores) ----
    dim3 gf(N_ITEMS / 16, BATCH / 16);
    final_gemm_kernel<<<gf, 32, 0, stream>>>(hbuf, batch_idxes, out);
}
